// Action_PureGRU_73349451481394
// MI455X (gfx1250) — matrix operand order repaired and checked by reading, not compiled, not run on hardware
//
#include <hip/hip_runtime.h>

#define BATCH 2048
#define FSTEPS 50
#define SDIM 1024
#define S3 3072
#define LN_EPS 1e-3f
#define UPD_BIAS -1.0f

typedef __attribute__((ext_vector_type(16))) __bf16          v16bf;
typedef __attribute__((ext_vector_type(8)))  float           v8f;
typedef __attribute__((ext_vector_type(8)))  unsigned short  v8us;
typedef __attribute__((ext_vector_type(16))) unsigned short  v16us;

struct U2 { v8us lo, hi; };

static __device__ __forceinline__ unsigned short f2bf(float f) {
    unsigned u = __float_as_uint(f);
    unsigned r = (u + 0x7fffu + ((u >> 16) & 1u)) >> 16;   // RNE
    return (unsigned short)r;
}

// ---------------- setup kernels ----------------

// W_combo[i][j] = sum_k W_emb[i][k] * W_gru[k][j]   (x-path collapsed to rank-2)
__global__ __launch_bounds__(256) void k_wcombo(const float* __restrict__ W_emb,
                                                const float* __restrict__ W_gru,
                                                float* __restrict__ Wc) {
    int idx = blockIdx.x * 256 + threadIdx.x;          // 2 * 3072
    if (idx >= 2 * S3) return;
    int i = idx / S3, j = idx - i * S3;
    float s = 0.f;
    for (int k = 0; k < SDIM; ++k)
        s += W_emb[i * SDIM + k] * W_gru[(size_t)k * S3 + j];
    Wc[idx] = s;
}

// W_hT[n][k] = bf16( W_gru[S + k][n] )   (h-path weights, transposed, bf16)
__global__ __launch_bounds__(256) void k_transpose(const float* __restrict__ W_gru,
                                                   unsigned short* __restrict__ WhT) {
    int idx = blockIdx.x * 256 + threadIdx.x;          // 3072 * 1024
    if (idx >= S3 * SDIM) return;
    int n = idx / SDIM, k = idx - n * SDIM;
    WhT[(size_t)n * SDIM + k] = f2bf(W_gru[(size_t)(SDIM + k) * S3 + n]);
}

__global__ __launch_bounds__(256) void k_zero_h(float* __restrict__ hf,
                                                unsigned short* __restrict__ hb) {
    int idx = blockIdx.x * 256 + threadIdx.x;
    if (idx >= BATCH * SDIM) return;
    hf[idx] = 0.f;
    hb[idx] = 0;
}

// ---------------- per-step GEMM: parts = h @ W_h + x_t @ W_combo ----------------
// block = 256 threads = 8 waves arranged 4(M) x 2(N); block tile 128 x 128;
// wave tile 32(M) x 64(N): 2 A-frags x 4 B-frags -> 8 WMMAs per K-step.
// Explicit 2-stage software pipeline over K (stride 64 = 2 x 32).

struct Frags {
    v16bf a0, a1;
    v16bf b[4];
};

static __device__ __forceinline__ void load_frags(Frags& f,
                                                  const unsigned short* aptr0,
                                                  const unsigned short* aptr1,
                                                  const unsigned short* bptr,
                                                  int k0) {
    U2 at;
    at.lo = *(const v8us*)(aptr0 + k0);
    at.hi = *(const v8us*)(aptr0 + k0 + 8);
    f.a0 = __builtin_bit_cast(v16bf, at);
    at.lo = *(const v8us*)(aptr1 + k0);
    at.hi = *(const v8us*)(aptr1 + k0 + 8);
    f.a1 = __builtin_bit_cast(v16bf, at);
#pragma unroll
    for (int ni = 0; ni < 4; ++ni)
        f.b[ni] = __builtin_bit_cast(v16bf, *(const v16us*)(bptr + k0 + ni * 16 * SDIM));
}

static __device__ __forceinline__ void wmma8(const Frags& f, v8f acc[2][4]) {
#pragma unroll
    for (int ni = 0; ni < 4; ++ni) {
        acc[0][ni] = __builtin_amdgcn_wmma_f32_16x16x32_bf16(false, f.a0, false, f.b[ni],
                                                             (short)0, acc[0][ni], false, false);
        acc[1][ni] = __builtin_amdgcn_wmma_f32_16x16x32_bf16(false, f.a1, false, f.b[ni],
                                                             (short)0, acc[1][ni], false, false);
    }
}

__global__ __launch_bounds__(256) void k_gru_gemm(const unsigned short* __restrict__ WhT, // [S3][S] bf16
                                                  const unsigned short* __restrict__ hb,  // [B][S]  bf16
                                                  const float* __restrict__ Wc,           // [2][S3]
                                                  const float* __restrict__ action,       // [B][F][2]
                                                  float* __restrict__ parts,              // [B][S3]
                                                  int t) {
    const int lane   = threadIdx.x & 31;
    const int wave   = threadIdx.x >> 5;
    const int wave_m = wave >> 1;            // 0..3
    const int wave_n = wave & 1;             // 0..1
    const int m0     = blockIdx.y * 128 + wave_m * 32;
    const int n0     = blockIdx.x * 128 + wave_n * 64;

    const int half  = lane >> 4;             // 0: lanes 0-15, 1: lanes 16-31
    const int l15   = lane & 15;
    const int kb_a  = half * 16;             // A frag: K offsets kb..kb+15 (contiguous)
    const int kb_b  = half * 16;             // B frag: K offsets kb..kb+15 (contiguous)

    v8f acc[2][4];
#pragma unroll
    for (int mi = 0; mi < 2; ++mi)
#pragma unroll
        for (int ni = 0; ni < 4; ++ni) acc[mi][ni] = (v8f){};

    const unsigned short* aptr0 = hb + (size_t)(m0 + l15) * SDIM + kb_a;
    const unsigned short* aptr1 = aptr0 + (size_t)16 * SDIM;
    const unsigned short* bptr  = WhT + (size_t)(n0 + l15) * SDIM + kb_b;

    Frags fA, fB;
    load_frags(fA, aptr0, aptr1, bptr, 0);                 // prologue: k = 0

    for (int k0 = 0; k0 < SDIM; k0 += 64) {
        load_frags(fB, aptr0, aptr1, bptr, k0 + 32);       // in-flight while computing fA
        wmma8(fA, acc);
        if (k0 + 64 < SDIM)
            load_frags(fA, aptr0, aptr1, bptr, k0 + 64);   // in-flight while computing fB
        wmma8(fB, acc);
    }

    // Epilogue: D layout -> row = m0 + mi*16 + half*8 + v, col = n0 + ni*16 + l15.
    // Fuse rank-2 x-contribution: a0*Wc[0][c] + a1*Wc[1][c].
    float wc0[4], wc1[4];
#pragma unroll
    for (int ni = 0; ni < 4; ++ni) {
        int c = n0 + ni * 16 + l15;
        wc0[ni] = Wc[c];
        wc1[ni] = Wc[S3 + c];
    }
#pragma unroll
    for (int mi = 0; mi < 2; ++mi) {
#pragma unroll
        for (int v = 0; v < 8; ++v) {
            int r = m0 + mi * 16 + half * 8 + v;
            float a0 = action[(size_t)r * (FSTEPS * 2) + t * 2 + 0];
            float a1 = action[(size_t)r * (FSTEPS * 2) + t * 2 + 1];
            float* prow = parts + (size_t)r * S3 + n0 + l15;
#pragma unroll
            for (int ni = 0; ni < 4; ++ni)
                prow[ni * 16] = acc[mi][ni][v] + a0 * wc0[ni] + a1 * wc1[ni];
        }
    }
}

// ---------------- per-step LayerNorm + gates (in-place h update) ----------------
__global__ __launch_bounds__(256) void k_gru_lngate(const float* __restrict__ parts,
                                                    const float* __restrict__ gamma,
                                                    const float* __restrict__ beta,
                                                    float* __restrict__ hf,
                                                    unsigned short* __restrict__ hb) {
    const int b = blockIdx.x;
    const float* row = parts + (size_t)b * S3;

    float s = 0.f, ss = 0.f;
    for (int j = threadIdx.x; j < S3; j += 256) {
        float x = row[j];
        s += x;
        ss += x * x;
    }
    __shared__ float sh1[256], sh2[256];
    sh1[threadIdx.x] = s;
    sh2[threadIdx.x] = ss;
    __syncthreads();
    for (int st = 128; st > 0; st >>= 1) {
        if (threadIdx.x < st) {
            sh1[threadIdx.x] += sh1[threadIdx.x + st];
            sh2[threadIdx.x] += sh2[threadIdx.x + st];
        }
        __syncthreads();
    }
    const float mu   = sh1[0] * (1.0f / S3);
    const float var  = sh2[0] * (1.0f / S3) - mu * mu;
    const float rstd = rsqrtf(var + LN_EPS);

    for (int j = threadIdx.x; j < SDIM; j += 256) {
        float rv = (row[j]            - mu) * rstd * gamma[j]            + beta[j];
        float cv = (row[SDIM + j]     - mu) * rstd * gamma[SDIM + j]     + beta[SDIM + j];
        float uv = (row[2 * SDIM + j] - mu) * rstd * gamma[2 * SDIM + j] + beta[2 * SDIM + j];
        float reset = 1.f / (1.f + __expf(-rv));
        float cand  = tanhf(reset * cv);
        float upd   = 1.f / (1.f + __expf(-(uv + UPD_BIAS)));
        size_t hi = (size_t)b * SDIM + j;
        float prev = hf[hi];
        float hn = upd * cand + (1.f - upd) * prev;
        hf[hi] = hn;
        hb[hi] = f2bf(hn);
    }
}

// ---------------- finalize: copy h_last + decoder ----------------
__global__ __launch_bounds__(256) void k_copy_h(const float* __restrict__ hf,
                                                float* __restrict__ out) {
    int idx = blockIdx.x * 256 + threadIdx.x;
    if (idx < BATCH * SDIM) out[idx] = hf[idx];
}

__global__ __launch_bounds__(256) void k_decode(const float* __restrict__ hf,
                                                const float* __restrict__ W_dec, // [S][2]
                                                const float* __restrict__ b_dec, // [2]
                                                float* __restrict__ out) {       // [B][2]
    int idx = blockIdx.x * 256 + threadIdx.x;            // BATCH * 2
    if (idx >= BATCH * 2) return;
    int b = idx >> 1, i = idx & 1;
    const float* hrow = hf + (size_t)b * SDIM;
    float s = 0.f;
    for (int k = 0; k < SDIM; ++k) s += hrow[k] * W_dec[k * 2 + i];
    out[idx] = s + b_dec[i];
}

// ---------------- launcher ----------------
extern "C" void kernel_launch(void* const* d_in, const int* in_sizes, int n_in,
                              void* d_out, int out_size, void* d_ws, size_t ws_size,
                              hipStream_t stream) {
    const float* action   = (const float*)d_in[0];  // [B, F, 2]
    const float* W_emb    = (const float*)d_in[1];  // [2, S]
    const float* W_gru    = (const float*)d_in[2];  // [2S, 3S]
    const float* ln_gamma = (const float*)d_in[3];  // [3S]
    const float* ln_beta  = (const float*)d_in[4];  // [3S]
    const float* W_dec    = (const float*)d_in[5];  // [S, 2]
    const float* b_dec    = (const float*)d_in[6];  // [2]

    char* ws = (char*)d_ws;
    size_t off = 0;
    unsigned short* WhT = (unsigned short*)(ws + off); off += (size_t)S3 * SDIM * 2;      // 6 MB
    float*          Wc  = (float*)(ws + off);          off += (size_t)2 * S3 * 4;          // 24 KB
    float*       parts  = (float*)(ws + off);          off += (size_t)BATCH * S3 * 4;      // 25 MB
    float*          hf  = (float*)(ws + off);          off += (size_t)BATCH * SDIM * 4;    // 8 MB
    unsigned short* hb  = (unsigned short*)(ws + off); off += (size_t)BATCH * SDIM * 2;    // 4 MB

    float* out_h   = (float*)d_out;                        // [B, 1, S]
    float* out_dec = (float*)d_out + (size_t)BATCH * SDIM; // [B, 1, 2]

    // setup
    k_wcombo   <<<(2 * S3 + 255) / 256, 256, 0, stream>>>(W_emb, W_gru, Wc);
    k_transpose<<<(S3 * SDIM + 255) / 256, 256, 0, stream>>>(W_gru, WhT);
    k_zero_h   <<<(BATCH * SDIM + 255) / 256, 256, 0, stream>>>(hf, hb);

    // sequential GRU scan
    dim3 gemm_grid(S3 / 128, BATCH / 128);   // 24 x 16
    for (int t = 0; t < FSTEPS; ++t) {
        k_gru_gemm  <<<gemm_grid, 256, 0, stream>>>(WhT, hb, Wc, action, parts, t);
        k_gru_lngate<<<BATCH, 256, 0, stream>>>(parts, ln_gamma, ln_beta, hf, hb);
    }

    // outputs
    k_copy_h<<<(BATCH * SDIM + 255) / 256, 256, 0, stream>>>(hf, out_h);
    k_decode<<<(BATCH * 2 + 255) / 256, 256, 0, stream>>>(hf, W_dec, b_dec, out_dec);
}
